// BoundaryLoss_5566277615923
// MI455X (gfx1250) — compile-verified, matrix operand order checked by reading, not run on hardware
//
#include <hip/hip_runtime.h>
#include <hip/hip_bf16.h>

typedef __attribute__((ext_vector_type(16))) _Float16 v16h;
typedef __attribute__((ext_vector_type(8)))  float    v8f;

#define HW 256
#define PIX 65536            // 256*256
#define NPIX_TOTAL 262144    // 4 * 65536
#define NBLK 32              // reduction blocks
#define RED_THREADS 8192     // NBLK * 256
#define PPT 32               // pixels per thread (2 WMMA accumulations)

// ds_swizzle group-of-32 encodings: offset = (xor<<10) | (or<<5) | and
#define SWZ_X16 0x401F
#define SWZ_X8  0x201F
#define SWZ_X4  0x101F
#define SWZ_X2  0x081F
#define SWZ_X1  0x041F

template <int CODE>
__device__ __forceinline__ float swz_add(float s) {
    return s + __int_as_float(__builtin_amdgcn_ds_swizzle(__float_as_int(s), CODE));
}

#if defined(__gfx1250__) && __has_builtin(__builtin_amdgcn_global_load_async_to_lds_b32)
#define HAVE_ASYNC_LDS 1
typedef __attribute__((address_space(1))) int* glb_i32p;
typedef __attribute__((address_space(3))) int* lds_i32p;
#endif

// ---------------------------------------------------------------------------
// S1: column pass. f is binary (0 / INF), so the column min-plus collapses to
// a 1D nearest-zero distance: forward + backward scan per column, squared.
// One thread per (batch, column). Writes squared distances in-place to ws.
// ---------------------------------------------------------------------------
__global__ void edt_cols_kernel(const int* __restrict__ target,
                                float* __restrict__ gpos,
                                float* __restrict__ gneg) {
    const int b = blockIdx.x;      // 4 blocks
    const int j = threadIdx.x;     // 256 threads = columns
    const int* t = target + b * PIX;
    float* gp = gpos + b * PIX;
    float* gn = gneg + b * PIX;

    float dp = 1.0e6f, dn = 1.0e6f;
    for (int i = 0; i < HW; ++i) {
        const int v = t[i * HW + j];
        dp = (v == 0) ? 0.0f : dp + 1.0f;   // nearest background (pos EDT zeros)
        dn = (v != 0) ? 0.0f : dn + 1.0f;   // nearest foreground (neg EDT zeros)
        gp[i * HW + j] = dp;
        gn[i * HW + j] = dn;
    }
    dp = 1.0e6f; dn = 1.0e6f;
    for (int i = HW - 1; i >= 0; --i) {
        const int v = t[i * HW + j];
        dp = (v == 0) ? 0.0f : dp + 1.0f;
        dn = (v != 0) ? 0.0f : dn + 1.0f;
        const float fp = fminf(gp[i * HW + j], dp);
        const float fn = fminf(gn[i * HW + j], dn);
        gp[i * HW + j] = fp * fp;           // squared column distance
        gn[i * HW + j] = fn * fn;
    }
}

// ---------------------------------------------------------------------------
// S2: row pass. True min-plus: D2[i,j] = min_j' g[i,j'] + (j-j')^2, rows
// staged in LDS (async global->LDS copy when the toolchain exposes it).
// Then posdis/negdis -> sdf with boundary zeroing.
// One block per (batch,row), one thread per column.
// ---------------------------------------------------------------------------
__global__ void edt_rows_sdf_kernel(const int* __restrict__ target,
                                    const float* __restrict__ gpos,
                                    const float* __restrict__ gneg,
                                    float* __restrict__ sdf) {
    const int bi = blockIdx.x;      // 1024 blocks
    const int b = bi >> 8;
    const int i = bi & 255;
    const int j = threadIdx.x;

    __shared__ float rp[HW];
    __shared__ float rn[HW];
    const int rowbase = (b * HW + i) * HW;

#if HAVE_ASYNC_LDS
    __builtin_amdgcn_global_load_async_to_lds_b32(
        (glb_i32p)(gpos + rowbase + j), (lds_i32p)&rp[j], 0, 0);
    __builtin_amdgcn_global_load_async_to_lds_b32(
        (glb_i32p)(gneg + rowbase + j), (lds_i32p)&rn[j], 0, 0);
  #if __has_builtin(__builtin_amdgcn_s_wait_asynccnt)
    __builtin_amdgcn_s_wait_asynccnt(0);
  #else
    asm volatile("s_wait_asynccnt 0x0" ::: "memory");
  #endif
#else
    rp[j] = gpos[rowbase + j];
    rn[j] = gneg[rowbase + j];
#endif
    __syncthreads();

    float mp = 1.0e18f, mn = 1.0e18f;
    for (int jp = 0; jp < HW; ++jp) {
        const float dj = (float)(j - jp);
        const float d2 = dj * dj;
        mp = fminf(mp, rp[jp] + d2);   // LDS broadcast read, conflict-free
        mn = fminf(mn, rn[jp] + d2);
    }

    float posdis = fmodf(floorf(sqrtf(mp)), 256.0f);
    float negdis = fmodf(floorf(sqrtf(mn)), 256.0f);
    float r = fmodf(negdis - posdis, 256.0f);
    if (r < 0.0f) r += 256.0f;

    // inner boundary, 4-connectivity, image border counts as background
    const int* t = target + b * PIX;
    const int v = t[i * HW + j];
    bool boundary = false;
    if (v != 0) {
        const int up = (i > 0)      ? t[(i - 1) * HW + j] : 0;
        const int dn = (i < HW - 1) ? t[(i + 1) * HW + j] : 0;
        const int lf = (j > 0)      ? t[i * HW + j - 1]   : 0;
        const int rt = (j < HW - 1) ? t[i * HW + j + 1]   : 0;
        boundary = (up == 0) | (dn == 0) | (lf == 0) | (rt == 0);
    }
    sdf[rowbase + j] = boundary ? 0.0f : r;
}

// ---------------------------------------------------------------------------
// R: softmax fold + WMMA reduction.
// Per pixel: w = (e1+e2+e3)/sum(e); value = w * sdf. Each lane packs 16 f16
// values into A; B = all-ones; two chained v_wmma_f32_16x16x32_f16
// accumulate 1024 values per wave into f32 row sums. Because B=ones
// replicates row sums across all 16 columns, one SWAPX16 ds_swizzle + add
// leaves the EXACT wave total in every lane. Deterministic block partials.
// ---------------------------------------------------------------------------
__global__ void reduce_wmma_kernel(const float* __restrict__ pred,
                                   const float* __restrict__ sdf,
                                   float* __restrict__ partials) {
    const int t = blockIdx.x * blockDim.x + threadIdx.x;   // 0..8191

    v16h bones;
    #pragma unroll
    for (int k = 0; k < 16; ++k) bones[k] = (_Float16)1.0f;

    v8f c = {};
    #pragma unroll
    for (int half = 0; half < 2; ++half) {
        v16h a;
        #pragma unroll
        for (int k = 0; k < 16; ++k) {
            const int p   = (half * 16 + k) * RED_THREADS + t;  // coalesced slice
            const int b   = p >> 16;
            const int rem = p & (PIX - 1);
            const float* pp = pred + (size_t)b * 4 * PIX + rem;
            if (half * 16 + k + 1 < PPT) {
                __builtin_prefetch(pp + RED_THREADS, 0, 0);     // global_prefetch_b8
            }
            const float p0 = pp[0];
            const float p1 = pp[PIX];
            const float p2 = pp[2 * PIX];
            const float p3 = pp[3 * PIX];
            const float mx = fmaxf(fmaxf(p0, p1), fmaxf(p2, p3));
            const float e0 = __expf(p0 - mx);
            const float e1 = __expf(p1 - mx);
            const float e2 = __expf(p2 - mx);
            const float e3 = __expf(p3 - mx);
            const float w  = (e1 + e2 + e3) / (e0 + e1 + e2 + e3);
            a[k] = (_Float16)(w * sdf[p]);
        }
        // C += A x ones : K-accumulation through the matrix pipe
        c = __builtin_amdgcn_wmma_f32_16x16x32_f16(
            /*neg_a=*/false, a, /*neg_b=*/false, bones,
            /*c_mod=*/(short)0, c, /*reuse_a=*/false, /*reuse_b=*/false);
    }

    float s = 0.0f;
    #pragma unroll
    for (int k = 0; k < 8; ++k) s += c[k];

    // lanes 0-15 hold rows 0-7 of a column, lanes 16-31 rows 8-15:
    // one SWAPX16 swizzle-add -> every lane holds the full wave sum (exact).
    s = swz_add<SWZ_X16>(s);

    __shared__ float wsums[8];
    const int lane = threadIdx.x & 31;
    const int wv   = threadIdx.x >> 5;
    if (lane == 0) wsums[wv] = s;
    __syncthreads();
    if (threadIdx.x == 0) {
        float bs = 0.0f;
        #pragma unroll
        for (int k = 0; k < 8; ++k) bs += wsums[k];
        partials[blockIdx.x] = bs;
    }
}

// ---------------------------------------------------------------------------
// F: one wave folds the 32 block partials into the scalar loss.
// mean over 4 batches x 3 channels x 256 x 256 = 786432 elements.
// ---------------------------------------------------------------------------
__global__ void finalize_kernel(const float* __restrict__ partials,
                                float* __restrict__ out) {
    float s = partials[threadIdx.x];          // 32 threads, 32 partials
    s = swz_add<SWZ_X16>(s);
    s = swz_add<SWZ_X8>(s);
    s = swz_add<SWZ_X4>(s);
    s = swz_add<SWZ_X2>(s);
    s = swz_add<SWZ_X1>(s);
    if (threadIdx.x == 0) {
        out[0] = s / 786432.0f;
    }
}

extern "C" void kernel_launch(void* const* d_in, const int* in_sizes, int n_in,
                              void* d_out, int out_size, void* d_ws, size_t ws_size,
                              hipStream_t stream) {
    const float* pred  = (const float*)d_in[0];   // (4,4,256,256) f32
    const int* target  = (const int*)d_in[1];     // (4,256,256) i32
    float* out = (float*)d_out;                   // scalar f32

    float* w = (float*)d_ws;
    float* gpos     = w;                    // 4*65536 floats
    float* gneg     = w + NPIX_TOTAL;       // 4*65536 floats
    float* sdfbuf   = w + 2 * NPIX_TOTAL;   // 4*65536 floats
    float* partials = w + 3 * NPIX_TOTAL;   // 32 floats

    edt_cols_kernel<<<4, 256, 0, stream>>>(target, gpos, gneg);
    edt_rows_sdf_kernel<<<1024, 256, 0, stream>>>(target, gpos, gneg, sdfbuf);
    reduce_wmma_kernel<<<NBLK, 256, 0, stream>>>(pred, sdfbuf, partials);
    finalize_kernel<<<1, 32, 0, stream>>>(partials, out);
}
